// LSTMDecoder_76587856822959
// MI455X (gfx1250) — matrix operand order repaired and checked by reading, not compiled, not run on hardware
//
#include <hip/hip_runtime.h>
#include <hip/hip_bf16.h>

// ---------------------------------------------------------------------------
// LSTM decoder w/ attention, teacher forcing.  S=256 B=64 D=512 T=64 V=32000
// Heavy math (gates, pe, logits GEMMs) runs on v_wmma_f32_16x16x32_bf16 with
// software-pipelined (double-buffered) fragment loads so VMEM overlaps the
// matrix pipe.  vocab table converted once to bf16 (32MB -> L2 resident).
// ---------------------------------------------------------------------------

typedef __bf16 bf16_t;
typedef __attribute__((ext_vector_type(16))) __bf16 v16bf;
typedef __attribute__((ext_vector_type(8)))  __bf16 v8bf;
typedef __attribute__((ext_vector_type(8)))  float  v8f;

constexpr int kS = 256;
constexpr int kB = 64;
constexpr int kD = 512;
constexpr int kT = 64;
constexpr int kV = 32000;
constexpr float kLnEps = 1e-5f;

union AB16 { v16bf v; v8bf h[2]; };

// A-matrix fragment (16x32 bf16, M x K), A row-major with leading dim lda.
// Layout: lanes 0-15 row M=lane hold K=0..15 in order; lanes 16-31 same
// rows hold K=16..31 (matches B fragment).
__device__ __forceinline__ v16bf load_a_frag(const bf16_t* A, int lda, int m0,
                                             int kb, int lane) {
  const int m  = m0 + (lane & 15);
  const int hi = (lane >> 4) & 1;
  return *(const v16bf*)(A + (size_t)m * lda + kb + hi * 16);
}

// B-matrix fragment (32x16 bf16, K x N) where B = W^T and W is row-major
// [N, K] with leading dim ldw.  Layout (same k order as A): lanes 0-15 col N=lane hold
// K=0..15 in order; lanes 16-31 hold K=16..31.  -> one contiguous 32B load.
__device__ __forceinline__ v16bf load_b_frag(const bf16_t* W, int ldw, int n0,
                                             int kb, int lane) {
  const int n  = n0 + (lane & 15);
  const int hi = (lane >> 4) & 1;
  return *(const v16bf*)(W + (size_t)n * ldw + kb + hi * 16);
}

__device__ __forceinline__ v8f wmma_bf16(v16bf a, v16bf b, v8f c) {
  return __builtin_amdgcn_wmma_f32_16x16x32_bf16(false, a, false, b,
                                                 (short)0, c, false, false);
}

// 64x16 output tile GEMM accumulator: acc[4] covers M=0..63, one 16-wide
// N tile at n0.  Double-buffered fragments: the k+1 loads are issued before
// the k-step WMMAs so s_wait_loadcnt becomes partial instead of a full drain.
template <int KTOT>
__device__ __forceinline__ void gemm_acc(v8f acc[4], const bf16_t* __restrict__ A,
                                         int lda, const bf16_t* __restrict__ W,
                                         int ldw, int n0, int lane) {
  v16bf bcur = load_b_frag(W, ldw, n0, 0, lane);
  v16bf acur[4];
#pragma unroll
  for (int mt = 0; mt < 4; ++mt) acur[mt] = load_a_frag(A, lda, mt * 16, 0, lane);

#pragma unroll 4
  for (int kb = 0; kb < KTOT - 32; kb += 32) {
    // prefetch next k-step fragments
    v16bf bnxt = load_b_frag(W, ldw, n0, kb + 32, lane);
    v16bf anxt[4];
#pragma unroll
    for (int mt = 0; mt < 4; ++mt)
      anxt[mt] = load_a_frag(A, lda, mt * 16, kb + 32, lane);
    // compute current k-step
#pragma unroll
    for (int mt = 0; mt < 4; ++mt) acc[mt] = wmma_bf16(acur[mt], bcur, acc[mt]);
    bcur = bnxt;
#pragma unroll
    for (int mt = 0; mt < 4; ++mt) acur[mt] = anxt[mt];
  }
  // peeled last k-step
#pragma unroll
  for (int mt = 0; mt < 4; ++mt) acc[mt] = wmma_bf16(acur[mt], bcur, acc[mt]);
}

__device__ __forceinline__ float sigmoidf(float x) {
  return 1.0f / (1.0f + __expf(-x));
}

// ---------------------------------------------------------------------------
// Prep kernels
// ---------------------------------------------------------------------------
__global__ void k_cvt_bf16(const float* __restrict__ src,
                           bf16_t* __restrict__ dst, int n) {
  int i = blockIdx.x * blockDim.x + threadIdx.x;
  if (i < n) dst[i] = (bf16_t)src[i];
}

// xs[t*B*D + b*D + d] = bf16(vocab[labels[b*T + t] * D + d])
__global__ void k_gather_emb(const int* __restrict__ labels,
                             const float* __restrict__ vocab,
                             bf16_t* __restrict__ xs) {
  int i = blockIdx.x * blockDim.x + threadIdx.x;     // over T*B*D
  if (i >= kT * kB * kD) return;
  int d  = i % kD;
  int bt = i / kD;
  int b  = bt % kB;
  int t  = bt / kB;
  int lab = labels[b * kT + t];
  xs[i] = (bf16_t)vocab[(size_t)lab * kD + d];
}

__global__ void k_init_state(const float* __restrict__ feats,
                             float* __restrict__ h, float* __restrict__ c,
                             bf16_t* __restrict__ hb) {
  int i = blockIdx.x * blockDim.x + threadIdx.x;     // over B*D
  if (i >= kB * kD) return;
  float v = feats[(size_t)(kS - 1) * kB * kD + i];
  h[i] = v;
  c[i] = v;
  hb[i] = (bf16_t)v;
}

// ---------------------------------------------------------------------------
// gates = x @ W_ih^T + h @ W_hh^T + b_ih + b_hh      [64, 2048]
// 128 waves; each wave owns one 16-col tile and all 4 row tiles (M=64).
// ---------------------------------------------------------------------------
__global__ __launch_bounds__(256) void k_gates(
    const bf16_t* __restrict__ x, const bf16_t* __restrict__ h,
    const bf16_t* __restrict__ Wih, const bf16_t* __restrict__ Whh,
    const float* __restrict__ b_ih, const float* __restrict__ b_hh,
    float* __restrict__ gates) {
  const int gtid = blockIdx.x * blockDim.x + threadIdx.x;
  const int wave = gtid >> 5;            // 0..127 -> n-tile
  const int lane = threadIdx.x & 31;
  const int n0 = wave * 16;

  v8f acc[4] = {};
  gemm_acc<kD>(acc, x, kD, Wih, kD, n0, lane);
  gemm_acc<kD>(acc, h, kD, Whh, kD, n0, lane);

  const int n  = n0 + (lane & 15);
  const int hi = (lane >> 4) & 1;
  const float bias = b_ih[n] + b_hh[n];
#pragma unroll
  for (int mt = 0; mt < 4; ++mt)
#pragma unroll
    for (int r = 0; r < 8; ++r)
      gates[(size_t)(mt * 16 + hi * 8 + r) * (4 * kD) + n] = acc[mt][r] + bias;
}

// ---------------------------------------------------------------------------
// LSTM cell elementwise
// ---------------------------------------------------------------------------
__global__ void k_cell(const float* __restrict__ gates, float* __restrict__ h,
                       float* __restrict__ c, bf16_t* __restrict__ hb) {
  int i = blockIdx.x * blockDim.x + threadIdx.x;     // over B*D
  if (i >= kB * kD) return;
  int b = i / kD, d = i % kD;
  const float* g = gates + (size_t)b * 4 * kD;
  float i_ = g[d];
  float f_ = g[kD + d];
  float g_ = g[2 * kD + d];
  float o_ = g[3 * kD + d];
  float cn = sigmoidf(f_) * c[i] + sigmoidf(i_) * tanhf(g_);
  float hn = sigmoidf(o_) * tanhf(cn);
  c[i] = cn;
  h[i] = hn;
  hb[i] = (bf16_t)hn;
}

// ---------------------------------------------------------------------------
// Attention: scores over S, softmax, context; writes hc = bf16([h, ctx]).
// One block per batch row, 256 threads (one per encoder position).
// ---------------------------------------------------------------------------
__global__ __launch_bounds__(256) void k_attn(const float* __restrict__ feats,
                                              const float* __restrict__ h,
                                              bf16_t* __restrict__ hc) {
  const int b = blockIdx.x;
  const int s = threadIdx.x;
  __shared__ float sm[kS];
  __shared__ float red[kS];

  const float* hrow = h + (size_t)b * kD;
  const float* frow = feats + (size_t)s * kB * kD + (size_t)b * kD;

  // dot(feats[s,b,:], h[b,:]) with 128-bit loads
  const float4* f4 = (const float4*)frow;
  const float4* h4 = (const float4*)hrow;
  float acc = 0.0f;
  for (int d = 0; d < kD / 4; ++d) {
    float4 fv = f4[d];
    float4 hv = h4[d];
    acc += fv.x * hv.x + fv.y * hv.y + fv.z * hv.z + fv.w * hv.w;
  }

  red[s] = acc;
  __syncthreads();
  for (int off = 128; off > 0; off >>= 1) {
    if (s < off) red[s] = fmaxf(red[s], red[s + off]);
    __syncthreads();
  }
  float mx = red[0];
  __syncthreads();

  float e = __expf(acc - mx);
  sm[s] = e;
  red[s] = e;
  __syncthreads();
  for (int off = 128; off > 0; off >>= 1) {
    if (s < off) red[s] += red[s + off];
    __syncthreads();
  }
  float inv = 1.0f / red[0];
  __syncthreads();

  for (int dd = 0; dd < kD; dd += 256) {
    int d = dd + s;
    float cacc = 0.0f;
    for (int ss = 0; ss < kS; ++ss)
      cacc += sm[ss] * feats[(size_t)ss * kB * kD + (size_t)b * kD + d];
    hc[(size_t)b * 2 * kD + kD + d] = (bf16_t)(cacc * inv);
    hc[(size_t)b * 2 * kD + d]      = (bf16_t)hrow[d];
  }
}

// ---------------------------------------------------------------------------
// pe_pre = hc @ W_hc^T + b_hc          [64, 512], K = 1024
// 32 waves (4 blocks x 8 waves).
// ---------------------------------------------------------------------------
__global__ __launch_bounds__(256) void k_pegemm(
    const bf16_t* __restrict__ hc, const bf16_t* __restrict__ Whc,
    const float* __restrict__ b_hc, float* __restrict__ pe_pre) {
  const int gtid = blockIdx.x * blockDim.x + threadIdx.x;
  const int wave = gtid >> 5;            // 0..31 -> n-tile
  const int lane = threadIdx.x & 31;
  const int n0 = wave * 16;

  v8f acc[4] = {};
  gemm_acc<2 * kD>(acc, hc, 2 * kD, Whc, 2 * kD, n0, lane);

  const int n  = n0 + (lane & 15);
  const int hi = (lane >> 4) & 1;
  const float bias = b_hc[n];
#pragma unroll
  for (int mt = 0; mt < 4; ++mt)
#pragma unroll
    for (int r = 0; r < 8; ++r)
      pe_pre[(size_t)(mt * 16 + hi * 8 + r) * kD + n] = acc[mt][r] + bias;
}

// ---------------------------------------------------------------------------
// pe = bf16(tanh(layernorm(pe_pre)))   one block per row, 256 threads.
// ---------------------------------------------------------------------------
__global__ __launch_bounds__(256) void k_ln_tanh(
    const float* __restrict__ pe_pre, const float* __restrict__ ln_g,
    const float* __restrict__ ln_b, bf16_t* __restrict__ pe) {
  const int b = blockIdx.x;
  const int t = threadIdx.x;
  __shared__ float red[256];

  float x0 = pe_pre[(size_t)b * kD + t];
  float x1 = pe_pre[(size_t)b * kD + 256 + t];

  red[t] = x0 + x1;
  __syncthreads();
  for (int off = 128; off > 0; off >>= 1) {
    if (t < off) red[t] += red[t + off];
    __syncthreads();
  }
  float mean = red[0] * (1.0f / kD);
  __syncthreads();

  float d0 = x0 - mean, d1 = x1 - mean;
  red[t] = d0 * d0 + d1 * d1;
  __syncthreads();
  for (int off = 128; off > 0; off >>= 1) {
    if (t < off) red[t] += red[t + off];
    __syncthreads();
  }
  float rstd = rsqrtf(red[0] * (1.0f / kD) + kLnEps);

  pe[(size_t)b * kD + t]       = (bf16_t)tanhf(ln_g[t] * d0 * rstd + ln_b[t]);
  pe[(size_t)b * kD + 256 + t] =
      (bf16_t)tanhf(ln_g[256 + t] * d1 * rstd + ln_b[256 + t]);
}

// ---------------------------------------------------------------------------
// logits = pe @ vocab^T + vocab_bias   [64, 32000], K = 512.  The hot GEMM:
// 2000 waves, each computes a 64x16 tile (B-frag reused over 4 M-tiles).
// ---------------------------------------------------------------------------
__global__ __launch_bounds__(256) void k_logits(
    const bf16_t* __restrict__ pe, const bf16_t* __restrict__ vocab,
    const float* __restrict__ vbias, float* __restrict__ out_t) {
  const int gtid = blockIdx.x * blockDim.x + threadIdx.x;
  const int wave = gtid >> 5;            // 0..1999 -> n-tile
  const int lane = threadIdx.x & 31;
  const int n0 = wave * 16;

  // touch this lane's vocab row (emits global_prefetch_b8; row is 1KB)
  {
    const bf16_t* row = vocab + (size_t)(n0 + (lane & 15)) * kD;
    __builtin_prefetch(row, 0, 1);
    __builtin_prefetch(row + kD / 2, 0, 1);
  }

  v8f acc[4] = {};
  gemm_acc<kD>(acc, pe, kD, vocab, kD, n0, lane);

  const int n  = n0 + (lane & 15);
  const int hi = (lane >> 4) & 1;
  const float bias = vbias[n];
#pragma unroll
  for (int mt = 0; mt < 4; ++mt)
#pragma unroll
    for (int r = 0; r < 8; ++r)
      out_t[(size_t)(mt * 16 + hi * 8 + r) * kV + n] = acc[mt][r] + bias;
}

// ---------------------------------------------------------------------------
// Host launch
// ---------------------------------------------------------------------------
extern "C" void kernel_launch(void* const* d_in, const int* in_sizes, int n_in,
                              void* d_out, int out_size, void* d_ws,
                              size_t ws_size, hipStream_t stream) {
  const float* feats  = (const float*)d_in[0];   // [S,B,D]
  const int*   labels = (const int*)d_in[1];     // [B,T]
  const float* W_ih   = (const float*)d_in[2];   // [4D,D]
  const float* W_hh   = (const float*)d_in[3];   // [4D,D]
  const float* b_ih   = (const float*)d_in[4];   // [4D]
  const float* b_hh   = (const float*)d_in[5];   // [4D]
  const float* W_hc   = (const float*)d_in[6];   // [D,2D]
  const float* b_hc   = (const float*)d_in[7];   // [D]
  const float* ln_g   = (const float*)d_in[8];   // [D]
  const float* ln_b   = (const float*)d_in[9];   // [D]
  const float* vocab  = (const float*)d_in[10];  // [V,D]
  const float* vbias  = (const float*)d_in[11];  // [V]
  float* out = (float*)d_out;                    // [T,B,V]

  char* ws = (char*)d_ws;
  size_t off = 0;
  auto alloc = [&](size_t bytes) -> void* {
    void* p = (void*)(ws + off);
    off += (bytes + 255) & ~(size_t)255;
    return p;
  };

  bf16_t* vocab_bf = (bf16_t*)alloc((size_t)kV * kD * 2);       // 32.8 MB
  bf16_t* Wih_bf   = (bf16_t*)alloc((size_t)4 * kD * kD * 2);   // 2 MB
  bf16_t* Whh_bf   = (bf16_t*)alloc((size_t)4 * kD * kD * 2);   // 2 MB
  bf16_t* Whc_bf   = (bf16_t*)alloc((size_t)kD * 2 * kD * 2);   // 1 MB
  bf16_t* xs_bf    = (bf16_t*)alloc((size_t)kT * kB * kD * 2);  // 4 MB
  float*  h_f      = (float*)alloc((size_t)kB * kD * 4);
  float*  c_f      = (float*)alloc((size_t)kB * kD * 4);
  bf16_t* h_bf     = (bf16_t*)alloc((size_t)kB * kD * 2);
  float*  gates_f  = (float*)alloc((size_t)kB * 4 * kD * 4);
  bf16_t* hc_bf    = (bf16_t*)alloc((size_t)kB * 2 * kD * 2);
  float*  pe_pre   = (float*)alloc((size_t)kB * kD * 4);
  bf16_t* pe_bf    = (bf16_t*)alloc((size_t)kB * kD * 2);
  (void)ws_size; (void)in_sizes; (void)n_in; (void)out_size;

  // ---- one-time prep -----------------------------------------------------
  {
    int n = kV * kD;
    k_cvt_bf16<<<(n + 255) / 256, 256, 0, stream>>>(vocab, vocab_bf, n);
    n = 4 * kD * kD;
    k_cvt_bf16<<<(n + 255) / 256, 256, 0, stream>>>(W_ih, Wih_bf, n);
    k_cvt_bf16<<<(n + 255) / 256, 256, 0, stream>>>(W_hh, Whh_bf, n);
    n = kD * 2 * kD;
    k_cvt_bf16<<<(n + 255) / 256, 256, 0, stream>>>(W_hc, Whc_bf, n);
    n = kT * kB * kD;
    k_gather_emb<<<(n + 255) / 256, 256, 0, stream>>>(labels, vocab, xs_bf);
    n = kB * kD;
    k_init_state<<<(n + 255) / 256, 256, 0, stream>>>(feats, h_f, c_f, h_bf);
  }

  // ---- sequential decode -------------------------------------------------
  for (int t = 0; t < kT; ++t) {
    k_gates<<<16, 256, 0, stream>>>(xs_bf + (size_t)t * kB * kD, h_bf, Wih_bf,
                                    Whh_bf, b_ih, b_hh, gates_f);
    k_cell<<<(kB * kD + 255) / 256, 256, 0, stream>>>(gates_f, h_f, c_f, h_bf);
    k_attn<<<kB, 256, 0, stream>>>(feats, h_f, hc_bf);
    k_pegemm<<<4, 256, 0, stream>>>(hc_bf, Whc_bf, b_hc, pe_pre);
    k_ln_tanh<<<kB, 256, 0, stream>>>(pe_pre, ln_g, ln_b, pe_bf);
    k_logits<<<250, 256, 0, stream>>>(pe_bf, vocab_bf, vbias,
                                      out + (size_t)t * kB * kV);
  }
}
